// PruMergeModule_68607807586425
// MI455X (gfx1250) — matrix operand order repaired and checked by reading, not compiled, not run on hardware
//
#include <hip/hip_runtime.h>

#define B_   8
#define P1_  4097
#define P_   4096
#define C_   1024
#define T_   256
#define NN_  3840
#define KP_  32

typedef __attribute__((ext_vector_type(16))) __bf16          v16bf;
typedef __attribute__((ext_vector_type(8)))  float           v8f;
typedef __attribute__((ext_vector_type(8)))  unsigned short  v8us;
typedef unsigned short u16t;

struct FragU { v8us lo; v8us hi; };   // 32 bytes, trivially copyable

__device__ __forceinline__ u16t f32_to_bf16(float f) {
  unsigned int x = __float_as_uint(f);
  unsigned int r = x + 0x7FFFu + ((x >> 16) & 1u);   // round-to-nearest-even
  return (u16t)(r >> 16);
}

// ---------------- K1: attn logits = (q_cls . k_p) / 32 ----------------
__global__ __launch_bounds__(256)
void attn_logits_kernel(const float* __restrict__ q, const float* __restrict__ k,
                        float* __restrict__ attn) {
  int p = blockIdx.x, b = blockIdx.y, tid = threadIdx.x;
  const float* qr = q + (size_t)b * P1_ * C_;
  const float* kr = k + ((size_t)b * P1_ + p) * C_;
  float s = 0.f;
  for (int c = tid; c < C_; c += 256) s += qr[c] * kr[c];
  __shared__ float red[256];
  red[tid] = s; __syncthreads();
  for (int off = 128; off > 0; off >>= 1) {
    if (tid < off) red[tid] += red[tid + off];
    __syncthreads();
  }
  if (tid == 0) attn[(size_t)b * P1_ + p] = red[0] * 0.03125f;  // C^-0.5
}

// ---------------- K2: softmax over P1 per batch (in place) ----------------
__global__ __launch_bounds__(256)
void softmax_kernel(float* __restrict__ attn) {
  int b = blockIdx.x, tid = threadIdx.x;
  float* a = attn + (size_t)b * P1_;
  __shared__ float red[256];
  float m = -3.402823e38f;
  for (int i = tid; i < P1_; i += 256) m = fmaxf(m, a[i]);
  red[tid] = m; __syncthreads();
  for (int off = 128; off > 0; off >>= 1) {
    if (tid < off) red[tid] = fmaxf(red[tid], red[tid + off]);
    __syncthreads();
  }
  float M = red[0]; __syncthreads();
  float s = 0.f;
  for (int i = tid; i < P1_; i += 256) s += __expf(a[i] - M);
  red[tid] = s; __syncthreads();
  for (int off = 128; off > 0; off >>= 1) {
    if (tid < off) red[tid] += red[tid + off];
    __syncthreads();
  }
  float inv = 1.f / red[0];
  for (int i = tid; i < P1_; i += 256) a[i] = __expf(a[i] - M) * inv;
}

// -------- K3: per-batch bitonic sort (desc, tie->low idx) + complement --------
__global__ __launch_bounds__(1024)
void topk_sort_kernel(const float* __restrict__ attn, int* __restrict__ topk_idx,
                      int* __restrict__ non_idx, float* __restrict__ non_attn) {
  int b = blockIdx.x, tid = threadIdx.x;
  __shared__ float sv[P_];
  __shared__ int   si[P_];
  __shared__ int   flg[P_];
  __shared__ int   scn[1024];
  const float* ca = attn + (size_t)b * P1_;
  for (int i = tid; i < P_; i += 1024) { sv[i] = ca[1 + i]; si[i] = i; }
  __syncthreads();
  for (int k = 2; k <= P_; k <<= 1) {
    for (int j = k >> 1; j > 0; j >>= 1) {
      for (int i = tid; i < P_; i += 1024) {
        int ixj = i ^ j;
        if (ixj > i) {
          float va = sv[i], vb = sv[ixj];
          int   ia = si[i], ib = si[ixj];
          bool asc = ((i & k) == 0);
          // precedence in desired order: larger value first, tie -> lower index
          bool precBA = (vb > va) || (vb == va && ib < ia);
          bool precAB = (va > vb) || (va == vb && ia < ib);
          if (asc ? precBA : precAB) {
            sv[i] = vb; sv[ixj] = va; si[i] = ib; si[ixj] = ia;
          }
        }
      }
      __syncthreads();
    }
  }
  if (tid < T_) topk_idx[b * T_ + tid] = si[tid];
  for (int i = tid; i < P_; i += 1024) flg[i] = 0;
  __syncthreads();
  if (tid < T_) flg[si[tid]] = 1;
  __syncthreads();
  int base4 = tid * 4;
  int s4 = 0;
  #pragma unroll
  for (int e = 0; e < 4; ++e) s4 += (flg[base4 + e] ^ 1);
  scn[tid] = s4; __syncthreads();
  for (int off = 1; off < 1024; off <<= 1) {
    int add = (tid >= off) ? scn[tid - off] : 0;
    __syncthreads();
    scn[tid] += add;
    __syncthreads();
  }
  int pos = scn[tid] - s4;   // exclusive prefix of unselected count
  for (int e = 0; e < 4; ++e) {
    int p = base4 + e;
    if (!flg[p]) {
      non_idx[b * NN_ + pos]  = p;
      non_attn[b * NN_ + pos] = ca[1 + p];
      pos++;
    }
  }
}

// -------- K4: gather keys by idx, L2-normalize, emit bf16 --------
__global__ __launch_bounds__(256)
void gather_norm_kernel(const float* __restrict__ k, const int* __restrict__ topk_idx,
                        const int* __restrict__ non_idx, u16t* __restrict__ tkeys,
                        u16t* __restrict__ nkeys) {
  int r = blockIdx.x, b = blockIdx.y, tid = threadIdx.x;
  int srcp; u16t* dst;
  if (r < T_) { srcp = topk_idx[b * T_ + r];        dst = tkeys + ((size_t)b * T_  + r)        * C_; }
  else        { srcp = non_idx[b * NN_ + (r - T_)]; dst = nkeys + ((size_t)b * NN_ + (r - T_)) * C_; }
  const float* src = k + ((size_t)b * P1_ + 1 + srcp) * C_;
  float v[4]; float ss = 0.f;
  #pragma unroll
  for (int e = 0; e < 4; ++e) { v[e] = src[tid + 256 * e]; ss += v[e] * v[e]; }
  __shared__ float red[256];
  red[tid] = ss; __syncthreads();
  for (int off = 128; off > 0; off >>= 1) {
    if (tid < off) red[tid] += red[tid + off];
    __syncthreads();
  }
  float inv = 1.f / fmaxf(sqrtf(red[0]), 1e-12f);
  #pragma unroll
  for (int e = 0; e < 4; ++e) dst[tid + 256 * e] = f32_to_bf16(v[e] * inv);
}

// -------- K5: sim = A(256x1024) x B(3840x1024)^T via v_wmma_f32_16x16x32_bf16 --------
__global__ __launch_bounds__(256)
void sim_wmma_kernel(const u16t* __restrict__ tkeys, const u16t* __restrict__ nkeys,
                     float* __restrict__ sim) {
  int mtile = blockIdx.x, b = blockIdx.y;
  int tid = threadIdx.x, lane = tid & 31, wave = tid >> 5;
  int mlo = lane & 15, hi = lane >> 4;
  __shared__ u16t Atile[16 * C_];             // 32 KB A tile in LDS
  {
    const v8us* src = (const v8us*)(tkeys + ((size_t)b * T_ + mtile * 16) * C_);
    v8us* dst = (v8us*)Atile;
    for (int i = tid; i < (16 * C_) / 8; i += 256) dst[i] = src[i];
  }
  __syncthreads();
  const u16t* Arow = Atile + mlo * C_;        // A: lane%16 -> M row
  int aoff = 8 * hi;                          // A: lanes16-31 start at K+8
  int boff = 8 * hi;                          // B: lanes16-31 start at K+8
  for (int nt = wave; nt < NN_ / 16; nt += 8) {
    const u16t* Brow = nkeys + ((size_t)b * NN_ + nt * 16 + mlo) * C_;  // column N = lane%16
    v8f acc = {0.f, 0.f, 0.f, 0.f, 0.f, 0.f, 0.f, 0.f};
    for (int kb = 0; kb < C_; kb += 32) {
      FragU fa, fb;
      fa.lo = *(const v8us*)(Arow + kb + aoff);
      fa.hi = *(const v8us*)(Arow + kb + 16 + aoff);
      fb.lo = *(const v8us*)(Brow + kb + boff);
      fb.hi = *(const v8us*)(Brow + kb + 16 + boff);
      v16bf av = __builtin_bit_cast(v16bf, fa);
      v16bf bv = __builtin_bit_cast(v16bf, fb);
      acc = __builtin_amdgcn_wmma_f32_16x16x32_bf16(false, av, false, bv,
                                                    (short)0, acc, false, false);
    }
    float* out = sim + ((size_t)b * T_ + mtile * 16) * NN_ + nt * 16;
    #pragma unroll
    for (int r = 0; r < 8; ++r) out[(size_t)(r + 8 * hi) * NN_ + mlo] = acc[r];
  }
}

// -------- K6: per (b,t): top-32 of sim row, weighted merge -> d_out --------
__global__ __launch_bounds__(256)
void merge_kernel(const float* __restrict__ feat, const float* __restrict__ sim,
                  const int* __restrict__ topk_idx, const int* __restrict__ non_idx,
                  const float* __restrict__ non_attn, float* __restrict__ out) {
  int t = blockIdx.x, b = blockIdx.y, tid = threadIdx.x;
  __shared__ float srow[NN_];
  __shared__ float rv[256];
  __shared__ int   ri[256];
  __shared__ int   selRow[KP_];
  __shared__ float selW[KP_];
  const float* sp = sim + ((size_t)b * T_ + t) * NN_;
  for (int i = tid; i < NN_; i += 256) srow[i] = sp[i];
  __syncthreads();
  for (int it = 0; it < KP_; ++it) {
    float bv = -3.402823e38f; int bi = 0x7fffffff;
    for (int i = tid; i < NN_; i += 256) {
      float v = srow[i];
      if (v > bv) { bv = v; bi = i; }            // strict > keeps lowest index
    }
    rv[tid] = bv; ri[tid] = bi;
    __syncthreads();
    for (int off = 128; off > 0; off >>= 1) {
      if (tid < off) {
        float v2 = rv[tid + off]; int i2 = ri[tid + off];
        if (v2 > rv[tid] || (v2 == rv[tid] && i2 < ri[tid])) { rv[tid] = v2; ri[tid] = i2; }
      }
      __syncthreads();
    }
    if (tid == 0) {
      int gj = ri[0];
      selRow[it] = 1 + non_idx[b * NN_ + gj];
      selW[it]   = non_attn[b * NN_ + gj];
      srow[gj]   = -3.402823e38f;               // mask for next iteration
    }
    __syncthreads();
  }
  float wsum = 0.f;
  #pragma unroll
  for (int j = 0; j < KP_; ++j) wsum += selW[j];
  float invw = 1.f / (wsum + 1e-6f);
  float acc[4] = {0.f, 0.f, 0.f, 0.f};
  for (int j = 0; j < KP_; ++j) {
    const float* fr = feat + ((size_t)b * P1_ + selRow[j]) * C_;
    float w = selW[j];
    #pragma unroll
    for (int e = 0; e < 4; ++e) acc[e] += w * fr[tid + 256 * e];
  }
  int trow = 1 + topk_idx[b * T_ + t];
  const float* tf = feat + ((size_t)b * P1_ + trow) * C_;
  float* op = out + ((size_t)b * (T_ + 1) + 1 + t) * C_;
  #pragma unroll
  for (int e = 0; e < 4; ++e) op[tid + 256 * e] = tf[tid + 256 * e] + acc[e] * invw;
}

// -------- K7: copy CLS feature row --------
__global__ __launch_bounds__(256)
void cls_copy_kernel(const float* __restrict__ feat, float* __restrict__ out) {
  int b = blockIdx.x, tid = threadIdx.x;
  const float* src = feat + (size_t)b * P1_ * C_;
  float* dst = out + (size_t)b * (T_ + 1) * C_;
  for (int c = tid; c < C_; c += 256) dst[c] = src[c];
}

extern "C" void kernel_launch(void* const* d_in, const int* in_sizes, int n_in,
                              void* d_out, int out_size, void* d_ws, size_t ws_size,
                              hipStream_t stream) {
  const float* feat = (const float*)d_in[0];
  const float* q    = (const float*)d_in[1];
  const float* k    = (const float*)d_in[2];
  float* out = (float*)d_out;
  (void)in_sizes; (void)n_in; (void)out_size; (void)ws_size;

  char* ws = (char*)d_ws;
  size_t off = 0;
  auto alloc = [&](size_t bytes) -> void* {
    void* p = ws + off;
    off = (off + bytes + 255) & ~(size_t)255;
    return p;
  };
  float* attn     = (float*)alloc((size_t)B_ * P1_ * 4);
  int*   topk_idx = (int*)  alloc((size_t)B_ * T_  * 4);
  int*   non_idx  = (int*)  alloc((size_t)B_ * NN_ * 4);
  float* non_attn = (float*)alloc((size_t)B_ * NN_ * 4);
  u16t*  tkeys    = (u16t*) alloc((size_t)B_ * T_  * C_ * 2);
  u16t*  nkeys    = (u16t*) alloc((size_t)B_ * NN_ * C_ * 2);
  float* sim      = (float*)alloc((size_t)B_ * T_  * NN_ * 4);

  attn_logits_kernel<<<dim3(P1_, B_), 256, 0, stream>>>(q, k, attn);
  softmax_kernel<<<dim3(B_), 256, 0, stream>>>(attn);
  topk_sort_kernel<<<dim3(B_), 1024, 0, stream>>>(attn, topk_idx, non_idx, non_attn);
  gather_norm_kernel<<<dim3(P_, B_), 256, 0, stream>>>(k, topk_idx, non_idx, tkeys, nkeys);
  sim_wmma_kernel<<<dim3(T_ / 16, B_), 256, 0, stream>>>(tkeys, nkeys, sim);
  merge_kernel<<<dim3(T_, B_), 256, 0, stream>>>(feat, sim, topk_idx, non_idx, non_attn, out);
  cls_copy_kernel<<<dim3(B_), 256, 0, stream>>>(feat, out);
}
